// STMHead1_21749714386961
// MI455X (gfx1250) — matrix operand order repaired and checked by reading, not compiled, not run on hardware
//
#include <hip/hip_runtime.h>

typedef __attribute__((ext_vector_type(16))) _Float16 v16h;
typedef __attribute__((ext_vector_type(8)))  _Float16 v8h;
typedef __attribute__((ext_vector_type(8)))  float    v8f;

#define HW 625
#define THW 15000
#define CATT 20
#define NB 8
#define COUT 256
#define INV_SQRT512 0.04419417382415922f

// ---------------------------------------------------------------------------
// Attention: one wave32 per (b, q). Online softmax over THW with 20 f32 accs,
// butterfly-merged across lanes; writes concat(mem, fq, zero-pad) as NHWC f16
// with 64 channels (40 real + 24 zero pad so conv1 K is a multiple of 32).
// ---------------------------------------------------------------------------
__global__ __launch_bounds__(256) void attn_kernel(
    const float* __restrict__ fm, const float* __restrict__ fq,
    _Float16* __restrict__ y)
{
    int lane = threadIdx.x & 31;
    int wid  = blockIdx.x * (blockDim.x >> 5) + (threadIdx.x >> 5);
    int b = wid / HW, q = wid % HW;
    if (b >= NB) return;

    float fqv[CATT];
    const float* fqp = fq + (size_t)b * CATT * HW + q;
#pragma unroll
    for (int c = 0; c < CATT; ++c) fqv[c] = fqp[(size_t)c * HW];

    float m = -1e30f, Z = 0.f;
    float acc[CATT];
#pragma unroll
    for (int c = 0; c < CATT; ++c) acc[c] = 0.f;

    const float* fmb = fm + (size_t)b * CATT * THW;
    for (int t = lane; t < THW; t += 32) {
        float v[CATT];
        float s = 0.f;
#pragma unroll
        for (int c = 0; c < CATT; ++c) { v[c] = fmb[(size_t)c * THW + t]; s += v[c] * fqv[c]; }
        s *= INV_SQRT512;
        if (s > m) {
            float sc = __expf(m - s);
            Z *= sc;
#pragma unroll
            for (int c = 0; c < CATT; ++c) acc[c] *= sc;
            m = s;
        }
        float e = __expf(s - m);
        Z += e;
#pragma unroll
        for (int c = 0; c < CATT; ++c) acc[c] += e * v[c];
    }
    for (int off = 16; off > 0; off >>= 1) {
        float m2 = __shfl_xor(m, off, 32);
        float Z2 = __shfl_xor(Z, off, 32);
        float nm = fmaxf(m, m2);
        float s1 = __expf(m - nm), s2 = __expf(m2 - nm);
        Z = Z * s1 + Z2 * s2;
#pragma unroll
        for (int c = 0; c < CATT; ++c) {
            float a2 = __shfl_xor(acc[c], off, 32);
            acc[c] = acc[c] * s1 + a2 * s2;
        }
        m = nm;
    }
    float invZ = 1.0f / Z;

    float v1 = 0.f, v2f = 0.f, v3 = 0.f;
#pragma unroll
    for (int c = 0; c < CATT; ++c) {
        v1  = (lane == c)        ? acc[c] * invZ : v1;   // ch lane    in [0,20)
        v2f = (lane == c + CATT) ? fqv[c]        : v2f;  // ch lane    in [20,32)
        v3  = (lane + 12 == c)   ? fqv[c]        : v3;   // ch lane+32 in [32,40)
    }
    _Float16* yp = y + ((size_t)b * HW + q) * 64;
    yp[lane]      = (_Float16)(v1 + v2f);
    yp[lane + 32] = (_Float16)v3;
}

// ---------------------------------------------------------------------------
// Weight repack: OIHW f32 -> f16 WMMA B-fragment layout.
// K ordering: k = r*CinPad + c. dst[id], id = ((kt*16+nt)*32+lane)*16 + j;
// K = kt*32 + (lane>>4)*16 + j ; n = nt*16 + (lane&15).
// ---------------------------------------------------------------------------
__global__ __launch_bounds__(256) void repack_w(
    const float* __restrict__ src, _Float16* __restrict__ dst,
    int Cin, int CinPad)
{
    int id = blockIdx.x * blockDim.x + threadIdx.x;
    int total = CinPad * 9 * 256;
    if (id >= total) return;
    int j    = id & 15;
    int lane = (id >> 4) & 31;
    int nt   = (id >> 9) & 15;
    int kt   = id >> 13;
    int K = kt * 32 + ((lane >> 4) << 4) + j;
    int n = nt * 16 + (lane & 15);
    int r = K / CinPad, c = K % CinPad;
    float v = 0.f;
    if (c < Cin) v = src[((size_t)n * Cin + c) * 9 + r];
    dst[id] = (_Float16)v;
}

// ---------------------------------------------------------------------------
// 3x3 conv (pad 1) as implicit GEMM on v_wmma_f32_16x16x32_f16.
//
// Block: 256 threads = 8 waves; tile M=32 spatial x N=256 out-channels.
// The whole receptive field of the 32 positions (<=5 image rows + 1 zero row,
// 126 rows x CINPAD ch) is staged to LDS ONCE via the gfx1250 async copy path
// (global_load_async_to_lds_b128 with per-lane swizzled LDS destinations,
// ASYNCcnt-tracked), zero rows via plain ds stores. 16B chunks are XOR-
// swizzled (chunk ^ (row&31)) to spread banks. The K loop is then pure
// ds_load_b128 (A) + coalesced global_load_b128 (B) + WMMA: no barriers,
// no divergence (OOB taps read the zero row via address select).
// Each wave: 2 m-tiles x 2 n-tiles = 4 WMMA per 32-wide k-chunk.
// ReLU is fused into the *producer* epilogue (STORE_RELU), so staging is a
// pure copy; the last conv layer stores raw values for the heads.
// ---------------------------------------------------------------------------
__device__ inline v16h load_a_swz(const _Float16* xs, int rowOff, int sw,
                                  int ctChunk, int b8) {
    const _Float16* p0 = xs + rowOff + (((ctChunk + 2 * b8)     ^ sw) << 3);
    const _Float16* p1 = xs + rowOff + (((ctChunk + 2 * b8 + 1) ^ sw) << 3);
    v8h lo = *(const v8h*)p0;
    v8h hi = *(const v8h*)p1;
    v16h a;
#pragma unroll
    for (int j = 0; j < 8; ++j) { a[j] = lo[j]; a[j + 8] = hi[j]; }
    return a;
}

template<int CINPAD, bool STORE_RELU>
__global__ __launch_bounds__(256) void conv3x3_wmma(
    const _Float16* __restrict__ x, const _Float16* __restrict__ bpack,
    const float* __restrict__ bias, _Float16* __restrict__ y)
{
    constexpr int CT    = CINPAD >> 5;   // 32-wide k-chunks per tap
    constexpr int CROW8 = CINPAD >> 3;   // 16B chunks per staged row
    __shared__ __align__(16) _Float16 Xs[126][256];   // 64,512 B

    const int t    = threadIdx.x;
    const int lane = t & 31;
    const int wave = t >> 5;
    const int mTile = blockIdx.x;                   // 0..19, 32 positions each
    const int b     = blockIdx.y;

    const int hbase = (mTile * 32) / 25;
    const int hlo   = hbase - 1;                    // first staged image row

    // ---- stage receptive field into LDS (rows: 0 = zeros, 1..125 = data) ---
    // Valid chunks: async copy global->LDS (per-lane swizzled LDS dest).
    // Invalid/zero chunks: plain ds_store of zeros.
    {
        const int TOT = 126 * CROW8;
        _Float16* xs0 = &Xs[0][0];
        for (int idx = t; idx < TOT; idx += 256) {
            int l  = idx / CROW8;                   // LDS row
            int c8 = (idx % CROW8) << 3;            // half index within row
            int sw = l & 31;
            _Float16* dstp = xs0 + l * 256 + (((c8 >> 3) ^ sw) << 3);
            const _Float16* srcp = nullptr;
            if (l > 0) {
                int d  = l - 1;
                int hd = hlo + d / 25;
                int wd = d % 25;
                if ((unsigned)hd < 25u)
                    srcp = x + ((size_t)b * HW + hd * 25 + wd) * CINPAD + c8;
            }
            if (srcp) {
                unsigned lds = (unsigned)(size_t)dstp;   // low 32b = LDS offset
                asm volatile("global_load_async_to_lds_b128 %0, %1, off"
                             :: "v"(lds), "v"(srcp) : "memory");
            } else {
                v8h z = {};
                *(v8h*)dstp = z;
            }
        }
        asm volatile("s_wait_asynccnt 0x0" ::: "memory");
    }
    __syncthreads();

    // ---- wave tiling -------------------------------------------------------
    const int nt0 = wave * 2, nt1 = wave * 2 + 1;   // n-tiles owned by wave
    const int M   = lane & 15;
    const int b8  = lane >> 4;                      // selects K halves 0..15 / 16..31

    // per-lane positions for the two m-tiles
    const int p0 = mTile * 32 + M;
    const int p1 = p0 + 16;
    const int ph0 = p0 / 25, pw0 = p0 % 25;
    const int ph1 = p1 / 25, pw1 = p1 % 25;

    v8f acc00 = {}, acc01 = {}, acc10 = {}, acc11 = {};
    const _Float16* xs = &Xs[0][0];

    for (int r = 0; r < 9; ++r) {
        const int dh = (r < 3) ? -1 : ((r < 6) ? 0 : 1);
        const int dw = r - 3 * (dh + 1) - 1;

        const bool ok0 = (p0 < HW) && ((unsigned)(ph0 + dh) < 25u) && ((unsigned)(pw0 + dw) < 25u);
        const bool ok1 = (p1 < HW) && ((unsigned)(ph1 + dh) < 25u) && ((unsigned)(pw1 + dw) < 25u);
        const int lidx0 = ok0 ? ((ph0 + dh - hlo) * 25 + (pw0 + dw) + 1) : 0;
        const int lidx1 = ok1 ? ((ph1 + dh - hlo) * 25 + (pw1 + dw) + 1) : 0;
        const int row0 = lidx0 * 256, sw0 = lidx0 & 31;
        const int row1 = lidx1 * 256, sw1 = lidx1 & 31;

        for (int ct = 0; ct < CT; ++ct) {
            v16h a0 = load_a_swz(xs, row0, sw0, ct * 4, b8);
            v16h a1 = load_a_swz(xs, row1, sw1, ct * 4, b8);

            const int kt = r * CT + ct;
            const _Float16* bp0 = bpack + (((size_t)kt * 16 + nt0) * 32 + lane) * 16;
            const _Float16* bp1 = bpack + (((size_t)kt * 16 + nt1) * 32 + lane) * 16;
            v16h bf0 = *(const v16h*)bp0;
            v16h bf1 = *(const v16h*)bp1;

            // prefetch next chunk's B fragments (clamped, high locality)
            const int ktn = (kt + 1 < 9 * CT) ? kt + 1 : kt;
            __builtin_prefetch(bpack + (((size_t)ktn * 16 + nt0) * 32 + lane) * 16, 0, 3);

            acc00 = __builtin_amdgcn_wmma_f32_16x16x32_f16(false, a0, false, bf0, (short)0, acc00, false, false);
            acc01 = __builtin_amdgcn_wmma_f32_16x16x32_f16(false, a0, false, bf1, (short)0, acc01, false, false);
            acc10 = __builtin_amdgcn_wmma_f32_16x16x32_f16(false, a1, false, bf0, (short)0, acc10, false, false);
            acc11 = __builtin_amdgcn_wmma_f32_16x16x32_f16(false, a1, false, bf1, (short)0, acc11, false, false);
        }
    }

    // ---- epilogue: bias (+ ReLU for producer layers) + f16 store (NHWC) ----
    const int n0 = nt0 * 16 + M;
    const int n1 = nt1 * 16 + M;
    const float bias0 = bias[n0];
    const float bias1 = bias[n1];
    _Float16* yb = y + (size_t)b * HW * COUT;
#pragma unroll
    for (int v = 0; v < 8; ++v) {
        const int mg0 = mTile * 32 + (b8 << 3) + v;   // m-tile 0 row
        const int mg1 = mg0 + 16;                     // m-tile 1 row
        float o00 = acc00[v] + bias0, o01 = acc01[v] + bias1;
        float o10 = acc10[v] + bias0, o11 = acc11[v] + bias1;
        if (STORE_RELU) {
            o00 = fmaxf(o00, 0.f); o01 = fmaxf(o01, 0.f);
            o10 = fmaxf(o10, 0.f); o11 = fmaxf(o11, 0.f);
        }
        if (mg0 < HW) {
            yb[(size_t)mg0 * COUT + n0] = (_Float16)o00;
            yb[(size_t)mg0 * COUT + n1] = (_Float16)o01;
        }
        if (mg1 < HW) {
            yb[(size_t)mg1 * COUT + n0] = (_Float16)o10;
            yb[(size_t)mg1 * COUT + n1] = (_Float16)o11;
        }
    }
}

// ---------------------------------------------------------------------------
// Heads: one wave per (b, q). 6 dot-products over 256 channels, shuffle
// reduce, then BN (running stats 0/1) + exp + bbox epilogue in f32.
// ---------------------------------------------------------------------------
__global__ __launch_bounds__(256) void head_kernel(
    const _Float16* __restrict__ clsf, const _Float16* __restrict__ regf,
    const float* __restrict__ wcls, const float* __restrict__ bcls,
    const float* __restrict__ gcls, const float* __restrict__ becls,
    const float* __restrict__ wctr, const float* __restrict__ bctr,
    const float* __restrict__ gctr, const float* __restrict__ bectr,
    const float* __restrict__ woff, const float* __restrict__ boff,
    const float* __restrict__ goff, const float* __restrict__ beoff,
    const float* __restrict__ si,   const float* __restrict__ bi,
    float* __restrict__ out)
{
    int lane = threadIdx.x & 31;
    int wid  = blockIdx.x * (blockDim.x >> 5) + (threadIdx.x >> 5);
    int b = wid / HW, q = wid % HW;
    if (b >= NB) return;

    const _Float16* cf = clsf + ((size_t)b * HW + q) * COUT;
    const _Float16* rf = regf + ((size_t)b * HW + q) * COUT;

    float acls = 0.f, actr = 0.f, ao0 = 0.f, ao1 = 0.f, ao2 = 0.f, ao3 = 0.f;
#pragma unroll
    for (int i = 0; i < 8; ++i) {
        int c = lane + (i << 5);
        float fc = (float)cf[c], fr = (float)rf[c];
        acls += fc * wcls[c];
        actr += fc * wctr[c];
        ao0  += fr * woff[c];
        ao1  += fr * woff[256 + c];
        ao2  += fr * woff[512 + c];
        ao3  += fr * woff[768 + c];
    }
    for (int off = 16; off > 0; off >>= 1) {
        acls += __shfl_xor(acls, off, 32);
        actr += __shfl_xor(actr, off, 32);
        ao0  += __shfl_xor(ao0,  off, 32);
        ao1  += __shfl_xor(ao1,  off, 32);
        ao2  += __shfl_xor(ao2,  off, 32);
        ao3  += __shfl_xor(ao3,  off, 32);
    }
    if (lane == 0) {
        const float inv = 0.99999500003749969f;     // 1/sqrt(1 + 1e-5)
        float cls = (acls + bcls[0]) * (gcls[0] * inv) + becls[0];
        float ctr = (actr + bctr[0]) * (gctr[0] * inv) + bectr[0];
        float s = si[0], bsh = bi[0];
        float o0 = expf(s * ((ao0 + boff[0]) * (goff[0] * inv) + beoff[0]) + bsh) * 8.f;
        float o1 = expf(s * ((ao1 + boff[1]) * (goff[1] * inv) + beoff[1]) + bsh) * 8.f;
        float o2 = expf(s * ((ao2 + boff[2]) * (goff[2] * inv) + beoff[2]) + bsh) * 8.f;
        float o3 = expf(s * ((ao3 + boff[3]) * (goff[3] * inv) + beoff[3]) + bsh) * 8.f;

        int idx = b * HW + q;
        out[idx]           = cls;
        out[NB * HW + idx] = ctr;
        float gx = 3.f + 8.f * (float)(q % 25);
        float gy = 3.f + 8.f * (float)(q / 25);
        float* ob = out + 2 * NB * HW + (size_t)idx * 4;
        ob[0] = gx - o0;
        ob[1] = gy - o1;
        ob[2] = gx + o2;
        ob[3] = gy + o3;
    }
}

// ---------------------------------------------------------------------------
extern "C" void kernel_launch(void* const* d_in, const int* in_sizes, int n_in,
                              void* d_out, int out_size, void* d_ws, size_t ws_size,
                              hipStream_t stream) {
    const float* fm     = (const float*)d_in[0];
    const float* fq     = (const float*)d_in[1];
    const float* cls1_w = (const float*)d_in[2];
    const float* cls1_b = (const float*)d_in[3];
    const float* clsk_w = (const float*)d_in[4];
    const float* clsk_b = (const float*)d_in[5];
    const float* reg1_w = (const float*)d_in[6];
    const float* reg1_b = (const float*)d_in[7];
    const float* regk_w = (const float*)d_in[8];
    const float* regk_b = (const float*)d_in[9];
    const float* w_cls  = (const float*)d_in[10];
    const float* b_cls  = (const float*)d_in[11];
    const float* g_cls  = (const float*)d_in[12];
    const float* be_cls = (const float*)d_in[13];
    const float* w_ctr  = (const float*)d_in[14];
    const float* b_ctr  = (const float*)d_in[15];
    const float* g_ctr  = (const float*)d_in[16];
    const float* be_ctr = (const float*)d_in[17];
    const float* w_off  = (const float*)d_in[18];
    const float* b_off  = (const float*)d_in[19];
    const float* g_off  = (const float*)d_in[20];
    const float* be_off = (const float*)d_in[21];
    const float* si     = (const float*)d_in[22];
    const float* bi     = (const float*)d_in[23];
    float* out = (float*)d_out;

    // workspace carve-up (all 256B aligned)
    char* ws = (char*)d_ws;
    _Float16* Y   = (_Float16*)(ws + 0);          // 8*625*64  f16 = 640,000 B
    _Float16* A1  = (_Float16*)(ws + 640000);     // 8*625*256 f16 = 2,560,000 B
    _Float16* A2  = (_Float16*)(ws + 3200000);
    _Float16* A3  = (_Float16*)(ws + 5760000);
    _Float16* A4  = (_Float16*)(ws + 8320000);
    _Float16* Bc1 = (_Float16*)(ws + 10880000);   // 576*256  f16 = 294,912 B
    _Float16* Br1 = (_Float16*)(ws + 11174912);
    _Float16* Bck = (_Float16*)(ws + 11469824);   // 6 * 2304*256 f16 = 7,077,888 B
    _Float16* Brk = (_Float16*)(ws + 18547712);   // total usage: 25,625,600 B

    const int LW32 = 256 * 256 * 9;      // f32 elems per 256->256 layer weight
    const int LB16 = 2304 * 256;         // f16 elems per repacked layer

    // 1) repack all conv weights into WMMA B-fragment layout
    repack_w<<<576, 256, 0, stream>>>(cls1_w, Bc1, 40, 64);
    repack_w<<<576, 256, 0, stream>>>(reg1_w, Br1, 40, 64);
    for (int i = 0; i < 6; ++i) {
        repack_w<<<2304, 256, 0, stream>>>(clsk_w + (size_t)i * LW32,
                                           Bck + (size_t)i * LB16, 256, 256);
        repack_w<<<2304, 256, 0, stream>>>(regk_w + (size_t)i * LW32,
                                           Brk + (size_t)i * LB16, 256, 256);
    }

    // 2) attention -> Y (NHWC f16, 64 channels incl. zero pad)
    attn_kernel<<<625, 256, 0, stream>>>(fm, fq, Y);

    // 3) conv towers (block tile M=32 x N=256, grid 20 x 8)
    //    ReLU is applied at the producer store of every layer except the last
    //    (heads consume the last layer's raw output).
    dim3 cg(20, NB);
    conv3x3_wmma<64, true><<<cg, 256, 0, stream>>>(Y, Bc1, cls1_b, A1);
    {
        _Float16* pin = A1; _Float16* pout = A2;
        for (int i = 0; i < 6; ++i) {
            if (i < 5)
                conv3x3_wmma<256, true><<<cg, 256, 0, stream>>>(
                    pin, Bck + (size_t)i * LB16, clsk_b + i * 256, pout);
            else
                conv3x3_wmma<256, false><<<cg, 256, 0, stream>>>(
                    pin, Bck + (size_t)i * LB16, clsk_b + i * 256, pout);
            _Float16* tmp = pin; pin = pout; pout = tmp;
        }   // cls_feat ends in A1
    }
    conv3x3_wmma<64, true><<<cg, 256, 0, stream>>>(Y, Br1, reg1_b, A3);
    {
        _Float16* pin = A3; _Float16* pout = A4;
        for (int i = 0; i < 6; ++i) {
            if (i < 5)
                conv3x3_wmma<256, true><<<cg, 256, 0, stream>>>(
                    pin, Brk + (size_t)i * LB16, regk_b + i * 256, pout);
            else
                conv3x3_wmma<256, false><<<cg, 256, 0, stream>>>(
                    pin, Brk + (size_t)i * LB16, regk_b + i * 256, pout);
            _Float16* tmp = pin; pin = pout; pout = tmp;
        }   // reg_feat ends in A3
    }

    // 4) heads + bbox epilogue
    head_kernel<<<625, 256, 0, stream>>>(A1, A3,
        w_cls, b_cls, g_cls, be_cls,
        w_ctr, b_ctr, g_ctr, be_ctr,
        w_off, b_off, g_off, be_off,
        si, bi, out);

    (void)in_sizes; (void)n_in; (void)out_size; (void)ws_size;
}
